// SelfAttention_6365141532722
// MI455X (gfx1250) — matrix operand order repaired and checked by reading, not compiled, not run on hardware
//
#include <hip/hip_runtime.h>
#include <hip/hip_bf16.h>
#include <stdint.h>

#define B_  4
#define S_  2048
#define D_  2048
#define H_  16
#define HD_ 128

typedef unsigned short u16;
typedef __attribute__((ext_vector_type(16))) __bf16 bf16x16;
typedef __attribute__((ext_vector_type(8)))  float  floatx8;

struct Frag { uint4 lo, hi; };   // 32 bytes = 16 bf16 elements

// ---- gfx1250 async-to-LDS availability -------------------------------------
#if __has_builtin(__builtin_amdgcn_global_load_async_to_lds_b128)
#define ASYNC_LDS 1
#else
#define ASYNC_LDS 0
#endif

#if __has_builtin(__builtin_amdgcn_s_wait_asynccnt)
#define WAIT_ASYNC(n) __builtin_amdgcn_s_wait_asynccnt(n)
#else
#define WAIT_ASYNC(n) asm volatile("s_wait_asynccnt %0" ::"n"(n) : "memory")
#endif

// exact parameter types per the builtin's signature:
//   arg0: int __vector(4) __device__ *  (global, non-const)
//   arg1: int __vector(4) __shared__ *  (LDS)
typedef int v4i_raw __attribute__((vector_size(16)));
typedef __attribute__((address_space(1))) v4i_raw* gvec_t;
typedef __attribute__((address_space(3))) v4i_raw* lvec_t;

__device__ __forceinline__ u16 f2bf(float f) {
  unsigned u = __float_as_uint(f);
  u += 0x7FFFu + ((u >> 16) & 1u);           // round-to-nearest-even
  return (u16)(u >> 16);
}
__device__ __forceinline__ float bf2f(u16 h) {
  return __uint_as_float(((unsigned)h) << 16);
}
__device__ __forceinline__ floatx8 wmma_bf16(const Frag& a, const Frag& b, floatx8 c) {
  return __builtin_amdgcn_wmma_f32_16x16x32_bf16(
      false, __builtin_bit_cast(bf16x16, a),
      false, __builtin_bit_cast(bf16x16, b),
      (short)0, c, false, false);
}

// ---------------------------------------------------------------- convert
__global__ __launch_bounds__(256) void f32_to_bf16_kernel(
    const float* __restrict__ in, u16* __restrict__ out, long long n) {
  long long i = (long long)blockIdx.x * 256 + threadIdx.x;
  if (i < n) out[i] = f2bf(in[i]);
}

// ---------------------------------------------------------------- GEMM
// C[M][N] = A[M][K] * B[N][K]^T   (A,B bf16 row-major, f32 accumulate)
// Block tile 128x128, K-step 32. 8 waves: 4 along M (32 rows), 2 along N (64 cols).
// Double-buffered LDS staging via GLOBAL_LOAD_ASYNC_TO_LDS_B128 (ASYNCcnt).
template <bool OUT_BF16>
__global__ __launch_bounds__(256) void gemm_bf16_kernel(
    const u16* __restrict__ A, const u16* __restrict__ B, void* __restrict__ C,
    int M, int N, int K) {
  constexpr int BM = 128, BN = 128, BK = 32;
  __shared__ __align__(16) u16 sA[2][BM * BK];
  __shared__ __align__(16) u16 sB[2][BN * BK];

  const int bm   = blockIdx.x * BM;
  const int bn   = blockIdx.y * BN;
  const int tid  = threadIdx.x;
  const int wave = tid >> 5;
  const int lane = tid & 31;
  const int half = lane >> 4;
  const int l16  = lane & 15;
  const int wm   = (wave & 3) * 32;   // wave M offset in block
  const int wn   = (wave >> 2) * 64;  // wave N offset in block

  floatx8 acc[2][4];
#pragma unroll
  for (int i = 0; i < 2; ++i)
#pragma unroll
    for (int j = 0; j < 4; ++j)
#pragma unroll
      for (int r = 0; r < 8; ++r) acc[i][j][r] = 0.0f;

  // issue one K-tile's staging (each thread: 2 chunks of A + 2 of B, b128 each)
  auto issue_tile = [&](int k0, int buf) {
#pragma unroll
    for (int i = 0; i < 2; ++i) {
      int idx = tid + i * 256;             // chunk id: row = idx>>2, kchunk = idx&3
      int r = idx >> 2, c = idx & 3;
#if ASYNC_LDS
      __builtin_amdgcn_global_load_async_to_lds_b128(
          (gvec_t)(A + (size_t)(bm + r) * K + k0 + c * 8),
          (lvec_t)(&sA[buf][idx * 8]), 0, 0);
      __builtin_amdgcn_global_load_async_to_lds_b128(
          (gvec_t)(B + (size_t)(bn + r) * K + k0 + c * 8),
          (lvec_t)(&sB[buf][idx * 8]), 0, 0);
#else
      ((uint4*)sA[buf])[idx] = *(const uint4*)(A + (size_t)(bm + r) * K + k0 + c * 8);
      ((uint4*)sB[buf])[idx] = *(const uint4*)(B + (size_t)(bn + r) * K + k0 + c * 8);
#endif
    }
  };

  const int nk = K / BK;
  issue_tile(0, 0);

  for (int kt = 0; kt < nk; ++kt) {
    const int buf = kt & 1;
    if (kt + 1 < nk) {
      issue_tile((kt + 1) * BK, (kt + 1) & 1);
#if ASYNC_LDS
      WAIT_ASYNC(4);    // in-order: oldest group (tile kt) complete
#endif
    } else {
#if ASYNC_LDS
      WAIT_ASYNC(0);
#endif
    }
    __syncthreads();    // all waves' tile-kt data visible in LDS

    const u16* cA = sA[buf];
    const u16* cB = sB[buf];
    Frag af[2], bf[4];
#pragma unroll
    for (int i = 0; i < 2; ++i) {          // A frag: lane<16 K0-15; lane>=16 K16-31
      const uint4* p = (const uint4*)(cA + (wm + i * 16 + l16) * BK);
      af[i].lo = p[half * 2];
      af[i].hi = p[half * 2 + 1];
    }
#pragma unroll
    for (int j = 0; j < 4; ++j) {          // B frag: lane<16 K0-15; lane>=16 K16-31
      const uint4* p = (const uint4*)(cB + (wn + j * 16 + l16) * BK);
      bf[j].lo = p[half * 2];
      bf[j].hi = p[half * 2 + 1];
    }
#pragma unroll
    for (int i = 0; i < 2; ++i)
#pragma unroll
      for (int j = 0; j < 4; ++j)
        acc[i][j] = wmma_bf16(af[i], bf[j], acc[i][j]);

    __syncthreads();    // done reading buf before it is overwritten (iter kt+2)
  }

  // epilogue: C layout -> lane holds col n = l16, rows r + 8*half
#pragma unroll
  for (int i = 0; i < 2; ++i)
#pragma unroll
    for (int j = 0; j < 4; ++j) {
      int col = bn + wn + j * 16 + l16;
#pragma unroll
      for (int r = 0; r < 8; ++r) {
        int row = bm + wm + i * 16 + r + 8 * half;
        if constexpr (OUT_BF16)
          ((u16*)C)[(size_t)row * N + col] = f2bf(acc[i][j][r]);
        else
          ((float*)C)[(size_t)row * N + col] = acc[i][j][r];
      }
    }
}

// ---------------------------------------------------------------- RoPE + split
// qkv (b,s,3*D) bf16 -> q,k (b,h,s,d) bf16 with rope, vT (b,h,d,s) bf16
__global__ __launch_bounds__(256) void rope_split_kernel(
    const u16* __restrict__ qkv, u16* __restrict__ q, u16* __restrict__ k,
    u16* __restrict__ vt) {
  long long gid = (long long)blockIdx.x * 256 + threadIdx.x;  // (b,h,s,j) j in [0,64)
  int j = (int)(gid & 63);
  long long t = gid >> 6;
  int s = (int)(t % S_); t /= S_;
  int h = (int)(t % H_);
  int b = (int)(t / H_);

  const size_t base = ((size_t)b * S_ + s) * (3 * D_) + (size_t)h * HD_;
  float q1 = bf2f(qkv[base + 2 * j]),           q2 = bf2f(qkv[base + 2 * j + 1]);
  float k1 = bf2f(qkv[base + D_ + 2 * j]),      k2 = bf2f(qkv[base + D_ + 2 * j + 1]);

  int i = (2 * j) & 63;                            // inv_freq index
  float inv = __expf(-9.2103403720f * (float)i * (1.0f / 64.0f)); // theta^(-i/64)
  float ang = (float)s * inv;
  float sn, cs;
  __sincosf(ang, &sn, &cs);

  size_t ob = (((size_t)b * H_ + h) * S_ + s) * HD_;
  q[ob + j]      = f2bf(q1 * cs - q2 * sn);
  q[ob + j + 64] = f2bf(q1 * sn + q2 * cs);
  k[ob + j]      = f2bf(k1 * cs - k2 * sn);
  k[ob + j + 64] = f2bf(k1 * sn + k2 * cs);

  size_t vb = ((size_t)b * H_ + h) * HD_;          // vT row base (d-major)
  vt[(vb + 2 * j)     * S_ + s] = qkv[base + 2 * D_ + 2 * j];
  vt[(vb + 2 * j + 1) * S_ + s] = qkv[base + 2 * D_ + 2 * j + 1];
}

// ---------------------------------------------------------------- attention
// one wave per 16-query tile; flash-style over 32-key iterations
__global__ __launch_bounds__(256) void attn_kernel(
    const u16* __restrict__ q, const u16* __restrict__ k,
    const u16* __restrict__ vt, u16* __restrict__ y) {
  __shared__ __align__(16) u16 sP[8][16 * 32];     // per-wave P staging (C->A layout)

  const int wave = threadIdx.x >> 5;
  const int lane = threadIdx.x & 31;
  const int half = lane >> 4;
  const int l16  = lane & 15;

  const int tile  = blockIdx.x * 8 + wave;  // 0..8191
  const int qt    = tile & 127;             // S/16 tiles
  const int bh    = tile >> 7;              // b*H + h
  const int qbase = qt * 16;
  const int b     = bh >> 4;
  const int h     = bh & 15;

  const u16* Q  = q  + (size_t)bh * S_ * HD_ + (size_t)qbase * HD_;
  const u16* Kp = k  + (size_t)bh * S_ * HD_;
  const u16* VT = vt + (size_t)bh * HD_ * S_;

  // Q fragments: 4 chunks of 16x32 over d
  Frag qf[4];
#pragma unroll
  for (int c = 0; c < 4; ++c) {
    const uint4* p = (const uint4*)(Q + (size_t)l16 * HD_ + c * 32);
    qf[c].lo = p[half * 2];
    qf[c].hi = p[half * 2 + 1];
  }

  floatx8 O[8];
  float mrow[8], lrow[8];
#pragma unroll
  for (int c = 0; c < 8; ++c)
#pragma unroll
    for (int r = 0; r < 8; ++r) O[c][r] = 0.0f;
#pragma unroll
  for (int r = 0; r < 8; ++r) { mrow[r] = -1e30f; lrow[r] = 0.0f; }

  const float scale = 0.08838834764831845f;   // 1/sqrt(128)
  const int np = qt / 2 + 1;                  // 32-key pair tiles (causal)

  for (int p = 0; p < np; ++p) {
    const int kb = p * 32;
    floatx8 s0, s1;
#pragma unroll
    for (int r = 0; r < 8; ++r) { s0[r] = 0.0f; s1[r] = 0.0f; }

    // scores: S[q, key] = sum_d Q K ; B frag lane holds key column, d contiguous
#pragma unroll
    for (int c = 0; c < 4; ++c) {
      Frag kf0, kf1;
      const uint4* p0 = (const uint4*)(Kp + (size_t)(kb + l16) * HD_ + c * 32 + half * 16);
      kf0.lo = p0[0]; kf0.hi = p0[1];
      const uint4* p1 = (const uint4*)(Kp + (size_t)(kb + 16 + l16) * HD_ + c * 32 + half * 16);
      kf1.lo = p1[0]; kf1.hi = p1[1];
      s0 = wmma_bf16(qf[c], kf0, s0);
      s1 = wmma_bf16(qf[c], kf1, s1);
    }

    float alpha[8];
#pragma unroll
    for (int r = 0; r < 8; ++r) {
      const int m = qbase + r + 8 * half;
      float a = s0[r] * scale;
      float bb = s1[r] * scale;
      if (kb + l16 > m)      a  = -1e30f;   // causal mask (only bites on last tile)
      if (kb + 16 + l16 > m) bb = -1e30f;
      float mx = fmaxf(a, bb);
      mx = fmaxf(mx, __shfl_xor(mx, 1));
      mx = fmaxf(mx, __shfl_xor(mx, 2));
      mx = fmaxf(mx, __shfl_xor(mx, 4));
      mx = fmaxf(mx, __shfl_xor(mx, 8));
      float mn = fmaxf(mrow[r], mx);
      alpha[r] = __expf(mrow[r] - mn);
      mrow[r] = mn;
      float p0 = __expf(a - mn), p1 = __expf(bb - mn);
      float rs = p0 + p1;
      rs += __shfl_xor(rs, 1);
      rs += __shfl_xor(rs, 2);
      rs += __shfl_xor(rs, 4);
      rs += __shfl_xor(rs, 8);
      lrow[r] = lrow[r] * alpha[r] + rs;
      // stage P to LDS (row-major 16x32, A-fragment friendly)
      sP[wave][(r + 8 * half) * 32 + l16]      = f2bf(p0);
      sP[wave][(r + 8 * half) * 32 + 16 + l16] = f2bf(p1);
    }

#pragma unroll
    for (int c = 0; c < 8; ++c)
#pragma unroll
      for (int r = 0; r < 8; ++r) O[c][r] *= alpha[r];

    // P as A fragment (same-wave LDS, in-order)
    Frag pf;
    {
      const uint4* pp = (const uint4*)(&sP[wave][l16 * 32]);
      pf.lo = pp[half * 2];
      pf.hi = pp[half * 2 + 1];
    }
    // accumulate P @ V : B frag from vT rows (d), keys contiguous
#pragma unroll
    for (int c = 0; c < 8; ++c) {
      Frag vf;
      const uint4* pv = (const uint4*)(VT + (size_t)(c * 16 + l16) * S_ + kb + half * 16);
      vf.lo = pv[0]; vf.hi = pv[1];
      O[c] = wmma_bf16(pf, vf, O[c]);
    }
  }

  // epilogue: y (b, s, h*128+d) bf16
#pragma unroll
  for (int c = 0; c < 8; ++c)
#pragma unroll
    for (int r = 0; r < 8; ++r) {
      int m = qbase + r + 8 * half;
      int d = c * 16 + l16;
      y[((size_t)b * S_ + m) * D_ + (size_t)h * HD_ + d] = f2bf(O[c][r] / lrow[r]);
    }
}

// ---------------------------------------------------------------- launch
extern "C" void kernel_launch(void* const* d_in, const int* in_sizes, int n_in,
                              void* d_out, int out_size, void* d_ws, size_t ws_size,
                              hipStream_t stream) {
  (void)in_sizes; (void)n_in; (void)out_size; (void)ws_size;
  const float* x     = (const float*)d_in[0];
  // d_in[1] = mask: causality computed analytically
  const float* w_qkv = (const float*)d_in[2];
  const float* w_out = (const float*)d_in[3];

  char* ws = (char*)d_ws;
  constexpr size_t XB_OFF   = 0;                                   // x bf16
  constexpr size_t WQKV_OFF = XB_OFF   + (size_t)B_ * S_ * D_ * 2; // w_qkv bf16
  constexpr size_t WOUT_OFF = WQKV_OFF + (size_t)3 * D_ * D_ * 2;  // w_out bf16
  constexpr size_t QKV_OFF  = WOUT_OFF + (size_t)D_ * D_ * 2;      // qkv bf16
  constexpr size_t Q_OFF    = QKV_OFF  + (size_t)B_ * S_ * 3 * D_ * 2;
  constexpr size_t K_OFF    = Q_OFF    + (size_t)B_ * S_ * D_ * 2;
  constexpr size_t VT_OFF   = K_OFF    + (size_t)B_ * S_ * D_ * 2;
  constexpr size_t Y_OFF    = VT_OFF   + (size_t)B_ * S_ * D_ * 2;

  u16* xb    = (u16*)(ws + XB_OFF);
  u16* wqkvb = (u16*)(ws + WQKV_OFF);
  u16* woutb = (u16*)(ws + WOUT_OFF);
  u16* qkvb  = (u16*)(ws + QKV_OFF);
  u16* qb    = (u16*)(ws + Q_OFF);
  u16* kb    = (u16*)(ws + K_OFF);
  u16* vtb   = (u16*)(ws + VT_OFF);
  u16* yb    = (u16*)(ws + Y_OFF);

  const long long nx = (long long)B_ * S_ * D_;
  const long long nq = (long long)3 * D_ * D_;
  const long long nw = (long long)D_ * D_;
  f32_to_bf16_kernel<<<(unsigned)((nx + 255) / 256), 256, 0, stream>>>(x, xb, nx);
  f32_to_bf16_kernel<<<(unsigned)((nq + 255) / 256), 256, 0, stream>>>(w_qkv, wqkvb, nq);
  f32_to_bf16_kernel<<<(unsigned)((nw + 255) / 256), 256, 0, stream>>>(w_out, woutb, nw);

  // QKV projection: M=8192, N=6144, K=2048 -> bf16 qkv
  gemm_bf16_kernel<true><<<dim3(8192 / 128, 6144 / 128), 256, 0, stream>>>(
      xb, wqkvb, qkvb, 8192, 6144, 2048);

  // RoPE + head split (+ V transpose)
  rope_split_kernel<<<(unsigned)(((long long)B_ * H_ * S_ * 64) / 256), 256, 0, stream>>>(
      qkvb, qb, kb, vtb);

  // Attention: B*H*(S/16)=8192 wave-tiles, 8 per block
  attn_kernel<<<1024, 256, 0, stream>>>(qb, kb, vtb, yb);

  // Output projection: M=8192, N=2048, K=2048 -> f32 out
  gemm_bf16_kernel<false><<<dim3(8192 / 128, 2048 / 128), 256, 0, stream>>>(
      yb, woutb, (float*)d_out, 8192, 2048, 2048);
}
